// ExpandedSchedule_59562606461423
// MI455X (gfx1250) — matrix operand order repaired and checked by reading, not compiled, not run on hardware
//
#include <hip/hip_runtime.h>
#include <hip/hip_bf16.h>
#include <math.h>

typedef __attribute__((ext_vector_type(16))) _Float16 v16h;
typedef __attribute__((ext_vector_type(8)))  _Float16 v8h;
typedef __attribute__((ext_vector_type(8)))  float    v8f;

#define NSTEPS   200000
#define ROWS     128         // time steps per workgroup in the MLP kernel
#define NBLOCKS  ((NSTEPS + ROWS - 1) / ROWS)   // 1563 (last block partial)
#define HID      256
#define CHUNK    500
#define NCHUNKS  400         // NCHUNKS * CHUNK == NSTEPS

__device__ __forceinline__ float gelu_exact(float x) {
  // jax.nn.gelu(approximate=False): 0.5*x*(1+erf(x/sqrt(2)))
  return 0.5f * x * (1.0f + erff(x * 0.70710678118654752440f));
}

__device__ __forceinline__ void write_row(float* __restrict__ out, int i,
                                          float beta, float kappa, float nu,
                                          float alpha, float lam) {
  float num = beta * lam * lam + nu * alpha * alpha - 2.0f * kappa * alpha * lam;
  float den = beta * nu - kappa * kappa;
  float* p = out + (size_t)i * 7;
  p[0] = alpha;            // mu[0]
  p[1] = lam;              // mu[1]
  p[2] = beta;             // cov flattened: beta,kappa,kappa,nu
  p[3] = kappa;
  p[4] = kappa;
  p[5] = nu;
  p[6] = logf(num) - logf(den);
}

// ---------------------------------------------------------------- kernel 1
// Convert fr_W2 (256x256 f32, row-major) to f16 once; it then lives in L2.
__global__ void prep_weights_kernel(const float* __restrict__ w2,
                                    _Float16* __restrict__ w2h) {
  int i = blockIdx.x * blockDim.x + threadIdx.x;
  if (i < HID * HID) w2h[i] = (_Float16)w2[i];
}

// ---------------------------------------------------------------- kernel 2
// Fused fr-MLP: t -> gelu(t*W1+b1) -> [WMMA] gelu(h@W2^T+b2) -> (f,r).
// Loop nest: kt (not unrolled) { load B(ct0), B(ct1); rt { load A once,
// 2 WMMAs } }. A read once from LDS, B once per kt; 16 live accumulators,
// ~170 VGPRs total -> no spills, no vgpr_msb.
__global__ __launch_bounds__(256) void mlp_wmma_kernel(
    const float* __restrict__ t_range,
    const float* __restrict__ W1, const float* __restrict__ b1,
    const float* __restrict__ b2,
    const float* __restrict__ W3, const float* __restrict__ b3,
    const _Float16* __restrict__ W2h,
    float4* __restrict__ params) {
  __shared__ __align__(32) _Float16 H1[ROWS][HID];   // row-major activations
  __shared__ __align__(32) _Float16 H2t[HID][ROWS];  // transposed layer-2 out

  const int tid  = threadIdx.x;
  const int row0 = blockIdx.x * ROWS;

  // ---- phase 1: layer 1 (K=1 outer product + exact GELU) into LDS, f16
  for (int idx = tid; idx < ROWS * HID; idx += 256) {
    int row = idx >> 8;
    int j   = idx & 255;
    int src = row0 + row;
    if (src > NSTEPS) src = NSTEPS;      // tail-block clamp (index valid)
    float t = t_range[src];
    H1[row][j] = (_Float16)gelu_exact(t * W1[j] + b1[j]);
  }
  __syncthreads();

  // ---- phase 2: layer 2 via v_wmma_f32_16x16x32_f16, register-blocked
  const int wave = tid >> 5;
  const int lane = tid & 31;
  const int half = lane >> 4;   // lane group (0: lanes 0-15, 1: lanes 16-31)
  const int ln   = lane & 15;

  const int nCol0 = wave * 16 + ln;            // column tile ct0 = wave
  const int nCol1 = (wave + 8) * 16 + ln;      // column tile ct1 = wave + 8

  v8f acc0[8], acc1[8];
#pragma unroll
  for (int rt = 0; rt < 8; ++rt) { acc0[rt] = (v8f){}; acc1[rt] = (v8f){}; }

#pragma unroll 1
  for (int kt = 0; kt < 8; ++kt) {
    const int kb = kt * 32;
    // B 32x16: B[k,n] = W2[n,k]; per lane k = kb + 16*half + e (contiguous)
    v16h b0 = *(const v16h*)&W2h[nCol0 * HID + kb + half * 16];
    v16h b1 = *(const v16h*)&W2h[nCol1 * HID + kb + half * 16];
#pragma unroll
    for (int rt = 0; rt < 8; ++rt) {
      // A 16x32 f16: per lane k = kb + 16*half + e (contiguous, matches B)
      const int rowA = rt * 16 + ln;
      v8h lo = *(const v8h*)&H1[rowA][kb + half * 16];
      v8h hi = *(const v8h*)&H1[rowA][kb + half * 16 + 8];
      v16h a;
#pragma unroll
      for (int e = 0; e < 8; ++e) { a[e] = lo[e]; a[e + 8] = hi[e]; }
      acc0[rt] = __builtin_amdgcn_wmma_f32_16x16x32_f16(
                     false, a, false, b0, (short)0, acc0[rt], false, false);
      acc1[rt] = __builtin_amdgcn_wmma_f32_16x16x32_f16(
                     false, a, false, b1, (short)0, acc1[rt], false, false);
    }
  }

  // C/D layout: element v -> (m = rt*16 + 8*half + v, n = nCol): the 8
  // elements are 8 consecutive rows -> one packed b128 store per tile.
  {
    const float bias0 = b2[nCol0];
    const float bias1 = b2[nCol1];
#pragma unroll
    for (int rt = 0; rt < 8; ++rt) {
      v8h hv0, hv1;
#pragma unroll
      for (int v = 0; v < 8; ++v) {
        hv0[v] = (_Float16)gelu_exact(acc0[rt][v] + bias0);
        hv1[v] = (_Float16)gelu_exact(acc1[rt][v] + bias1);
      }
      *(v8h*)&H2t[nCol0][rt * 16 + half * 8] = hv0;
      *(v8h*)&H2t[nCol1][rt * 16 + half * 8] = hv1;
    }
  }
  __syncthreads();

  // ---- phase 3: layer 3 (256 -> 2) per row, VALU, from transposed H2
  if (tid < ROWS) {
    const int row  = tid;
    const int step = row0 + row;
    if (step < NSTEPS) {
      float accf = 0.0f, accr = 0.0f;
      for (int k = 0; k < HID; ++k) {
        float h = (float)H2t[k][row];
        accf += h * W3[k];          // W3 row 0 -> f
        accr += h * W3[HID + k];    // W3 row 1 -> r
      }
      float f  = accf + b3[0];
      float r  = accr + b3[1];
      float dt = t_range[step + 1] - t_range[step];
      params[step] = make_float4(r, f, 0.0f, dt);
    }
  }
}

// ---------------------------------------------------------------- kernel 3
// Per-chunk product of transforms: 3x3 block (beta,kappa,nu) and 2x2 (alpha,lam).
__global__ void chunkprod_kernel(const float4* __restrict__ params,
                                 float* __restrict__ chP) {
  int c = blockIdx.x * blockDim.x + threadIdx.x;
  if (c >= NCHUNKS) return;
  float p00 = 1, p01 = 0, p02 = 0;
  float p10 = 0, p11 = 1, p12 = 0;
  float p20 = 0, p21 = 0, p22 = 1;
  float q00 = 1, q01 = 0, q10 = 0, q11 = 1;
  int base = c * CHUNK;
  for (int i = 0; i < CHUNK; ++i) {
    float4 pr = params[base + i];
    float r = pr.x, f = pr.y, dt = pr.w;
    // P_new = (I + dt*M) * P, rows of M3: [0,-r,0],[2,-f,-2r],[0,1,-2f]
    float n00 = p00 - dt * r * p10;
    float n01 = p01 - dt * r * p11;
    float n02 = p02 - dt * r * p12;
    float n10 = p10 + dt * (2.0f * p00 - f * p10 - 2.0f * r * p20);
    float n11 = p11 + dt * (2.0f * p01 - f * p11 - 2.0f * r * p21);
    float n12 = p12 + dt * (2.0f * p02 - f * p12 - 2.0f * r * p22);
    float n20 = p20 + dt * (p10 - 2.0f * f * p20);
    float n21 = p21 + dt * (p11 - 2.0f * f * p21);
    float n22 = p22 + dt * (p12 - 2.0f * f * p22);
    p00 = n00; p01 = n01; p02 = n02;
    p10 = n10; p11 = n11; p12 = n12;
    p20 = n20; p21 = n21; p22 = n22;
    // M2 rows: [0,-r],[1,-f]
    float m00 = q00 - dt * r * q10;
    float m01 = q01 - dt * r * q11;
    float m10 = q10 + dt * (q00 - f * q10);
    float m11 = q11 + dt * (q01 - f * q11);
    q00 = m00; q01 = m01; q10 = m10; q11 = m11;
  }
  float* o = chP + c * 16;
  o[0] = p00; o[1] = p01; o[2] = p02;
  o[3] = p10; o[4] = p11; o[5] = p12;
  o[6] = p20; o[7] = p21; o[8] = p22;
  o[9] = q00; o[10] = q01; o[11] = q10; o[12] = q11;
}

// ---------------------------------------------------------------- kernel 4
// Serial sweep over chunk products -> chunk-boundary states; emits row 0.
__global__ void boundary_kernel(const float* __restrict__ chP,
                                const float* __restrict__ log_beta_nu_zero,
                                const float* __restrict__ log_rho_zero,
                                float* __restrict__ bound,
                                float* __restrict__ out) {
  float beta  = expf(log_beta_nu_zero[0]);
  float nu    = expf(log_beta_nu_zero[1]);
  float rho   = 1.0f / (1.0f + expf(-log_rho_zero[0]));
  float kappa = rho * sqrtf(beta) * sqrtf(nu);
  float alpha = 1.0f, lam = 0.0f;

  write_row(out, 0, beta, kappa, nu, alpha, lam);

  for (int c = 0; c < NCHUNKS; ++c) {
    float* b = bound + c * 8;
    b[0] = beta; b[1] = kappa; b[2] = nu; b[3] = alpha; b[4] = lam;
    const float* p = chP + c * 16;
    float nb = p[0] * beta + p[1] * kappa + p[2] * nu;
    float nk = p[3] * beta + p[4] * kappa + p[5] * nu;
    float nn = p[6] * beta + p[7] * kappa + p[8] * nu;
    float na = p[9]  * alpha + p[10] * lam;
    float nl = p[11] * alpha + p[12] * lam;
    beta = nb; kappa = nk; nu = nn; alpha = na; lam = nl;
  }
}

// ---------------------------------------------------------------- kernel 5
// Each chunk re-propagates its 5-state recurrence and emits output rows.
__global__ void emit_kernel(const float4* __restrict__ params,
                            const float* __restrict__ bound,
                            float* __restrict__ out) {
  int c = blockIdx.x * blockDim.x + threadIdx.x;
  if (c >= NCHUNKS) return;
  const float* b = bound + c * 8;
  float beta = b[0], kappa = b[1], nu = b[2], alpha = b[3], lam = b[4];
  int base = c * CHUNK;
  for (int i = 0; i < CHUNK; ++i) {
    float4 pr = params[base + i];
    float r = pr.x, f = pr.y, dt = pr.w;
    float nb = beta  - dt * r * kappa;
    float nk = kappa + dt * (2.0f * beta - f * kappa - 2.0f * r * nu);
    float nn = nu    + dt * (kappa - 2.0f * f * nu);
    float na = alpha - dt * r * lam;
    float nl = lam   + dt * (alpha - f * lam);
    beta = nb; kappa = nk; nu = nn; alpha = na; lam = nl;
    write_row(out, base + i + 1, beta, kappa, nu, alpha, lam);
  }
}

// ---------------------------------------------------------------- launcher
extern "C" void kernel_launch(void* const* d_in, const int* in_sizes, int n_in,
                              void* d_out, int out_size, void* d_ws, size_t ws_size,
                              hipStream_t stream) {
  (void)in_sizes; (void)n_in; (void)out_size; (void)ws_size;
  const float* t_range = (const float*)d_in[0];
  const float* fr_W1   = (const float*)d_in[1];
  const float* fr_b1   = (const float*)d_in[2];
  const float* fr_W2   = (const float*)d_in[3];
  const float* fr_b2   = (const float*)d_in[4];
  const float* fr_W3   = (const float*)d_in[5];
  const float* fr_b3   = (const float*)d_in[6];
  // d_in[7..12]: g-MLP params — provably dead in the reference output (state 5
  // never feeds states 0-4 and is never emitted), so skipped entirely.
  const float* lbn     = (const float*)d_in[13];
  const float* lrho    = (const float*)d_in[14];
  float* out = (float*)d_out;

  // workspace layout (all offsets 256B-aligned)
  char* ws = (char*)d_ws;
  _Float16* w2h    = (_Float16*)ws;                                   // 131072 B
  float4*   params = (float4*)(ws + 131072);                          // 3.2 MB
  float*    chP    = (float*)(ws + 131072 + (size_t)NSTEPS * 16);     // 25600 B
  float*    bound  = chP + NCHUNKS * 16;                              // 12800 B

  prep_weights_kernel<<<(HID * HID + 255) / 256, 256, 0, stream>>>(fr_W2, w2h);

  mlp_wmma_kernel<<<NBLOCKS, 256, 0, stream>>>(
      t_range, fr_W1, fr_b1, fr_b2, fr_W3, fr_b3, w2h, params);

  chunkprod_kernel<<<(NCHUNKS + 255) / 256, 256, 0, stream>>>(params, chP);

  boundary_kernel<<<1, 1, 0, stream>>>(chP, lbn, lrho, bound, out);

  emit_kernel<<<(NCHUNKS + 255) / 256, 256, 0, stream>>>(params, bound, out);
}
